// ConcatCritic_31207232372863
// MI455X (gfx1250) — matrix operand order repaired and checked by reading, not compiled, not run on hardware
//
#include <hip/hip_runtime.h>
#include <hip/hip_bf16.h>

typedef __attribute__((ext_vector_type(16))) _Float16 v16h;
typedef __attribute__((ext_vector_type(8)))  float    v8f;

#define BSZ 512   // batch
#define HD  512   // hidden
#define DXY 128   // dx == dy

// ---------------------------------------------------------------------------
// Prep 1: Ax[i][k] = x[i] @ W1[0:128, k] ; Ay[j][k] = y[j] @ W1[128:256, k] + b1[k]
// 524288 threads, each a 128-length dot. Coalesced over k (consecutive tids).
// ---------------------------------------------------------------------------
__global__ __launch_bounds__(256) void prep_axy_kernel(
    const float* __restrict__ x, const float* __restrict__ y,
    const float* __restrict__ W1, const float* __restrict__ b1,
    float* __restrict__ Ax, float* __restrict__ Ay)
{
    int idx = blockIdx.x * 256 + threadIdx.x;          // 0 .. 2*512*512-1
    int sel = idx >> 18;                               // 0 -> Ax, 1 -> Ay
    int e   = idx & (BSZ * HD - 1);
    int row = e >> 9;                                  // 0..511
    int col = e & (HD - 1);                            // 0..511

    const float* vec = (sel == 0) ? (x + row * DXY) : (y + row * DXY);
    const float* wcol = W1 + (sel == 0 ? 0 : DXY) * HD + col;

    float sum = 0.0f;
#pragma unroll 8
    for (int d = 0; d < DXY; ++d)
        sum += vec[d] * wcol[(size_t)d * HD];

    if (sel == 0) Ax[e] = sum;
    else          Ay[e] = sum + b1[col];
}

// ---------------------------------------------------------------------------
// Prep 2: W2T[n][k] = (f16) W2[k][n]   (LDS-tiled transpose + convert)
// ---------------------------------------------------------------------------
__global__ __launch_bounds__(256) void prep_w2t_kernel(
    const float* __restrict__ W2, _Float16* __restrict__ W2T)
{
    __shared__ float tile[16][17];
    int tx = threadIdx.x & 15;
    int ty = threadIdx.x >> 4;
    int tileX = blockIdx.x & 31;   // n tile
    int tileY = blockIdx.x >> 5;   // k tile

    int k = tileY * 16 + ty;
    int n = tileX * 16 + tx;
    tile[ty][tx] = W2[(size_t)k * HD + n];            // coalesced in n
    __syncthreads();

    int nn = tileX * 16 + ty;
    int kk = tileY * 16 + tx;
    W2T[(size_t)nn * HD + kk] = (_Float16)tile[tx][ty]; // coalesced in k
}

// ---------------------------------------------------------------------------
// Main: per wave, 16 pairs (row i, cols jbase..jbase+15).
//   h1 = relu(Ax[i] + Ay[j])   (built as 16 WMMA A-fragments, f16, in VGPRs)
//   h2 = relu(h1 @ W2 + b2)    (v_wmma_f32_16x16x32_f16, f32 accum)
//   s  = h2 @ W3 + b3          (fused epilogue + lane butterfly reduction)
// ---------------------------------------------------------------------------
__global__ __launch_bounds__(128) void pair_mlp_kernel(
    const float* __restrict__ Ax, const float* __restrict__ Ay,
    const _Float16* __restrict__ W2T,
    const float* __restrict__ b2, const float* __restrict__ W3,
    const float* __restrict__ b3, float* __restrict__ out)
{
    const int lane = threadIdx.x & 31;
    const int wv   = (blockIdx.x << 2) + (threadIdx.x >> 5);  // global wave id
    const int i     = wv >> 5;               // output row (x index)
    const int jbase = (wv & 31) << 4;        // first output col (y index)
    const int m = lane & 15;                 // A-row / D-col within fragment
    const int g = lane >> 4;                 // lane half-group

    const float* axr = Ax + (size_t)i * HD;
    const float* ayr = Ay + (size_t)(jbase + m) * HD;

    // Build 16 A fragments (16x32 f16). Layout: group0 holds K{0-15},
    // group1 holds K{16-31} for each 32-wide K block (contiguous, matches B).
    v16h afrag[16];
#pragma unroll
    for (int kb = 0; kb < 16; ++kb) {
        int c0 = kb * 32 + g * 16;    // halves 0..7
        int c1 = c0 + 8;              // halves 8..15
        float4 a0 = *(const float4*)(axr + c0);
        float4 a1 = *(const float4*)(axr + c0 + 4);
        float4 p0 = *(const float4*)(ayr + c0);
        float4 p1 = *(const float4*)(ayr + c0 + 4);
        float4 a2 = *(const float4*)(axr + c1);
        float4 a3 = *(const float4*)(axr + c1 + 4);
        float4 p2 = *(const float4*)(ayr + c1);
        float4 p3 = *(const float4*)(ayr + c1 + 4);
        v16h f;
        f[0]  = (_Float16)fmaxf(a0.x + p0.x, 0.0f);
        f[1]  = (_Float16)fmaxf(a0.y + p0.y, 0.0f);
        f[2]  = (_Float16)fmaxf(a0.z + p0.z, 0.0f);
        f[3]  = (_Float16)fmaxf(a0.w + p0.w, 0.0f);
        f[4]  = (_Float16)fmaxf(a1.x + p1.x, 0.0f);
        f[5]  = (_Float16)fmaxf(a1.y + p1.y, 0.0f);
        f[6]  = (_Float16)fmaxf(a1.z + p1.z, 0.0f);
        f[7]  = (_Float16)fmaxf(a1.w + p1.w, 0.0f);
        f[8]  = (_Float16)fmaxf(a2.x + p2.x, 0.0f);
        f[9]  = (_Float16)fmaxf(a2.y + p2.y, 0.0f);
        f[10] = (_Float16)fmaxf(a2.z + p2.z, 0.0f);
        f[11] = (_Float16)fmaxf(a2.w + p2.w, 0.0f);
        f[12] = (_Float16)fmaxf(a3.x + p3.x, 0.0f);
        f[13] = (_Float16)fmaxf(a3.y + p3.y, 0.0f);
        f[14] = (_Float16)fmaxf(a3.z + p3.z, 0.0f);
        f[15] = (_Float16)fmaxf(a3.w + p3.w, 0.0f);
        afrag[kb] = f;
    }

    float partial[8];
#pragma unroll
    for (int r = 0; r < 8; ++r) partial[r] = 0.0f;

    // 32 column tiles of 16; 16 WMMAs (K=512) per tile.
    for (int nt = 0; nt < 32; ++nt) {
        int n = nt * 16 + m;                         // this lane's output column
        // B fragment (32x16 f16): lane group g holds K = 16*g..16*g+15, contiguous.
        const v16h* bptr = (const v16h*)(W2T + (size_t)n * HD + g * 16);
        v8f acc = {};
#pragma unroll
        for (int kb = 0; kb < 16; ++kb) {
            v16h bf = bptr[kb * 2];                  // advance 32 halves per K-block
            acc = __builtin_amdgcn_wmma_f32_16x16x32_f16(
                false, afrag[kb], false, bf, (short)0, acc, false, false);
        }
        float bias = b2[n];
        float w3v  = W3[n];
#pragma unroll
        for (int r = 0; r < 8; ++r) {
            float v = acc[r] + bias;                 // h2 pre-activation, row r+8g, col n
            v = fmaxf(v, 0.0f);
            partial[r] += v * w3v;
        }
    }

    // Reduce each partial over the 16 lanes of its half-group.
#pragma unroll
    for (int r = 0; r < 8; ++r) {
        float v = partial[r];
        v += __shfl_xor(v, 1);
        v += __shfl_xor(v, 2);
        v += __shfl_xor(v, 4);
        v += __shfl_xor(v, 8);
        partial[r] = v;
    }

    if (m == 0) {
        float bb = b3[0];
        float* orow = out + (size_t)i * BSZ + jbase + g * 8;  // rows r+8g -> cols
#pragma unroll
        for (int r = 0; r < 8; ++r) orow[r] = partial[r] + bb;
    }
}

// ---------------------------------------------------------------------------
extern "C" void kernel_launch(void* const* d_in, const int* in_sizes, int n_in,
                              void* d_out, int out_size, void* d_ws, size_t ws_size,
                              hipStream_t stream)
{
    const float* x  = (const float*)d_in[0];
    const float* y  = (const float*)d_in[1];
    const float* W1 = (const float*)d_in[2];
    const float* b1 = (const float*)d_in[3];
    const float* W2 = (const float*)d_in[4];
    const float* b2 = (const float*)d_in[5];
    const float* W3 = (const float*)d_in[6];
    const float* b3 = (const float*)d_in[7];
    float* out = (float*)d_out;

    char* ws = (char*)d_ws;
    float*    Ax  = (float*)(ws);                          // 1 MB
    float*    Ay  = (float*)(ws + (size_t)BSZ * HD * 4);   // 1 MB
    _Float16* W2T = (_Float16*)(ws + (size_t)2 * BSZ * HD * 4); // 0.5 MB

    prep_axy_kernel<<<(2 * BSZ * HD) / 256, 256, 0, stream>>>(x, y, W1, b1, Ax, Ay);
    prep_w2t_kernel<<<(HD / 16) * (HD / 16), 256, 0, stream>>>(W2, W2T);

    // 16384 waves, 16 pairs each; 4 waves (same i) per block.
    pair_mlp_kernel<<<4096, 128, 0, stream>>>(Ax, Ay, W2T, b2, W3, b3, out);
}
